// CausalSelfAttention_53489522704592
// MI455X (gfx1250) — matrix operand order repaired and checked by reading, not compiled, not run on hardware
//
#include <hip/hip_runtime.h>
#include <hip/hip_bf16.h>
#include <stdint.h>

// ---------------------------------------------------------------------------
// CausalSelfAttention forward for MI455X (gfx1250, wave32, WMMA bf16)
// B=4, T=2048, C=1024, H=16, D=64
// ---------------------------------------------------------------------------

typedef __bf16 bf16;
typedef __attribute__((ext_vector_type(16))) __bf16 v16bf;
typedef __attribute__((ext_vector_type(8)))  __bf16 v8bf;
typedef __attribute__((ext_vector_type(8)))  float  v8f;
typedef __attribute__((ext_vector_type(4)))  int    v4i;

#define B_   4
#define T_   2048
#define C_   1024
#define H_   16
#define D_   64

#define LDA_P 40   // padded LDS stride (elements); 80B rows keep 16B alignment
#define LDB_P 40

// -------- CDNA5 async global->LDS (ASYNCcnt) with safe fallback ------------
#if defined(__has_builtin)
#if __has_builtin(__builtin_amdgcn_global_load_async_to_lds_b128) && \
    __has_builtin(__builtin_amdgcn_s_wait_asynccnt)
#define USE_ASYNC_LDS 1
#endif
#endif

#if USE_ASYNC_LDS
typedef __attribute__((address_space(1))) v4i* g_v4i_ptr;
typedef __attribute__((address_space(3))) v4i* s_v4i_ptr;
#endif

__device__ __forceinline__ void copy16_g2s(const bf16* g, bf16* s) {
#if USE_ASYNC_LDS
    __builtin_amdgcn_global_load_async_to_lds_b128(
        (g_v4i_ptr)(void*)g, (s_v4i_ptr)(void*)s, 0, 0);
#else
    *reinterpret_cast<v8bf*>(s) = *reinterpret_cast<const v8bf*>(g);
#endif
}

__device__ __forceinline__ void copy_wait() {
#if USE_ASYNC_LDS
    __builtin_amdgcn_s_wait_asynccnt(0);
#endif
}

__device__ __forceinline__ bf16 f2bf(float f) {
    uint32_t u = __builtin_bit_cast(uint32_t, f);
    uint32_t r = u + 0x7FFFu + ((u >> 16) & 1u);   // round-to-nearest-even
    uint16_t h = (uint16_t)(r >> 16);
    return __builtin_bit_cast(bf16, h);
}

// ---------------------------------------------------------------------------
// f32 -> bf16 conversion, 4 elements per thread
// ---------------------------------------------------------------------------
__global__ void cvt4_kernel(const float* __restrict__ s, bf16* __restrict__ d, int n4) {
    int i = blockIdx.x * blockDim.x + threadIdx.x;
    if (i < n4) {
        const float4 v = reinterpret_cast<const float4*>(s)[i];
        uint64_t packed =
            ((uint64_t)__builtin_bit_cast(uint16_t, f2bf(v.x))      ) |
            ((uint64_t)__builtin_bit_cast(uint16_t, f2bf(v.y)) << 16) |
            ((uint64_t)__builtin_bit_cast(uint16_t, f2bf(v.z)) << 32) |
            ((uint64_t)__builtin_bit_cast(uint16_t, f2bf(v.w)) << 48);
        reinterpret_cast<uint64_t*>(d)[i] = packed;
    }
}

// ---------------------------------------------------------------------------
// Shared GEMM mainloop: block tile 128(M) x 128(N), K-step 32, 8 waves.
// Wave grid 4(M) x 2(N): each wave 32x64 = 2 A-frags x 4 B-frags = 8 WMMA/Kstep.
// A rows [Mtot, 1024] bf16 row-major; Bw [1024, NTOT] bf16 row-major.
// ---------------------------------------------------------------------------
template<int NTOT>
__device__ __forceinline__ void gemm_mainloop(
    const bf16* __restrict__ A, const bf16* __restrict__ Bw,
    bf16* As, bf16* Bt, int mBase, int nBase, v8f (&acc)[2][4])
{
    const int tid  = threadIdx.x;
    const int lane = tid & 31;
    const int wave = tid >> 5;
    const int hl   = lane >> 4;
    const int l16  = lane & 15;

    const int mSub = (wave & 3) * 32;
    const int nSub = (wave >> 2) * 64;

    const int arow = tid >> 1;            // 0..127
    const int acol = (tid & 1) * 16;      // 0 / 16
    const int brow = tid >> 3;            // 0..31
    const int bcol = (tid & 7) * 16;      // 0..112

    for (int kt = 0; kt < C_; kt += 32) {
        __syncthreads();
        // prefetch next K-step rows into cache hierarchy
        if (kt + 32 < C_) {
            __builtin_prefetch(&A[(size_t)(mBase + arow) * C_ + kt + 32 + acol], 0, 3);
            __builtin_prefetch(&Bw[(size_t)(kt + 32 + brow) * NTOT + nBase + bcol], 0, 3);
        }
        {   // A tile (row-major) -> LDS, async per-lane 16B copies
            const bf16* g = &A[(size_t)(mBase + arow) * C_ + kt + acol];
            bf16* sp = &As[arow * LDA_P + acol];
            copy16_g2s(g, sp);
            copy16_g2s(g + 8, sp + 8);
        }
        {   // B tile -> LDS transposed: Bt[n][k]
            const v8bf g0 = *reinterpret_cast<const v8bf*>(&Bw[(size_t)(kt + brow) * NTOT + nBase + bcol]);
            const v8bf g1 = *reinterpret_cast<const v8bf*>(&Bw[(size_t)(kt + brow) * NTOT + nBase + bcol + 8]);
            #pragma unroll
            for (int j = 0; j < 8; ++j) Bt[(bcol + j) * LDB_P + brow] = g0[j];
            #pragma unroll
            for (int j = 0; j < 8; ++j) Bt[(bcol + 8 + j) * LDB_P + brow] = g1[j];
        }
        copy_wait();
        __syncthreads();

        v16bf af[2], bfr[4];
        #pragma unroll
        for (int mi = 0; mi < 2; ++mi) {
            const bf16* p = &As[(mSub + mi * 16 + l16) * LDA_P + 8 * hl];
            v8bf lo = *reinterpret_cast<const v8bf*>(p);
            v8bf hi = *reinterpret_cast<const v8bf*>(p + 16);
            #pragma unroll
            for (int e = 0; e < 8; ++e) { af[mi][e] = lo[e]; af[mi][e + 8] = hi[e]; }
        }
        #pragma unroll
        for (int ni = 0; ni < 4; ++ni) {
            const bf16* p = &Bt[(nSub + ni * 16 + l16) * LDB_P + 8 * hl];
            v8bf lo = *reinterpret_cast<const v8bf*>(p);
            v8bf hi = *reinterpret_cast<const v8bf*>(p + 16);
            #pragma unroll
            for (int e = 0; e < 8; ++e) { bfr[ni][e] = lo[e]; bfr[ni][e + 8] = hi[e]; }
        }
        #pragma unroll
        for (int mi = 0; mi < 2; ++mi)
            #pragma unroll
            for (int ni = 0; ni < 4; ++ni)
                acc[mi][ni] = __builtin_amdgcn_wmma_f32_16x16x32_bf16(
                    false, af[mi], false, bfr[ni], (short)0, acc[mi][ni], false, false);
    }
}

// ---------------------------------------------------------------------------
// GEMM: qkv = x_bf @ w_qkv_bf, scatter into q[B,H,T,D], k[B,H,T,D], vt[B,H,D,T]
// ---------------------------------------------------------------------------
__global__ __launch_bounds__(256) void gemm_qkv_kernel(
    const bf16* __restrict__ A,    // [8192, 1024]
    const bf16* __restrict__ Bw,   // [1024, 3072]
    bf16* __restrict__ q, bf16* __restrict__ k, bf16* __restrict__ vt)
{
    __shared__ __align__(16) bf16 As[128 * LDA_P];
    __shared__ __align__(16) bf16 Bt[128 * LDB_P];

    const int lane = threadIdx.x & 31;
    const int wave = threadIdx.x >> 5;
    const int hl   = lane >> 4;
    const int l16  = lane & 15;

    const int nTiles = (3 * C_) / 128;            // 24
    const int mBase  = (blockIdx.x / nTiles) * 128;
    const int nBase  = (blockIdx.x % nTiles) * 128;

    v8f acc[2][4] = {};
    gemm_mainloop<3 * C_>(A, Bw, As, Bt, mBase, nBase, acc);

    const int mSub = (wave & 3) * 32;
    const int nSub = (wave >> 2) * 64;

    // Epilogue: scatter to Q / K / V^T (bf16). 128-wide N tile sits in one
    // segment (Q/K/V) and spans two heads.
    const int seg = nBase / C_;               // 0:Q 1:K 2:V
    const int bi  = mBase / T_;
    const int tB  = mBase % T_;

    #pragma unroll
    for (int mi = 0; mi < 2; ++mi) {
        #pragma unroll
        for (int ni = 0; ni < 4; ++ni) {
            const int cw = (nBase % C_) + nSub + ni * 16 + l16;  // 0..1023
            const int h  = cw >> 6;
            const int d  = cw & 63;
            #pragma unroll
            for (int r = 0; r < 8; ++r) {
                const int t = tB + mSub + mi * 16 + r + 8 * hl;
                bf16 val = f2bf(acc[mi][ni][r]);
                if (seg == 0)
                    q[(((size_t)bi * H_ + h) * T_ + t) * D_ + d] = val;
                else if (seg == 1)
                    k[(((size_t)bi * H_ + h) * T_ + t) * D_ + d] = val;
                else
                    vt[(((size_t)bi * H_ + h) * D_ + d) * (size_t)T_ + t] = val;
            }
        }
    }
}

// ---------------------------------------------------------------------------
// Flash attention: one wave handles 16 query rows of one (b,h).
// ---------------------------------------------------------------------------
__global__ __launch_bounds__(256) void attn_kernel(
    const bf16* __restrict__ q, const bf16* __restrict__ k,
    const bf16* __restrict__ vt, bf16* __restrict__ y)
{
    __shared__ __align__(16) bf16 pstage[8][16 * LDA_P];

    const int lane = threadIdx.x & 31;
    const int wave = threadIdx.x >> 5;
    const int hl   = lane >> 4;
    const int l16  = lane & 15;

    const int task = blockIdx.x * 8 + wave;   // 0..8191
    const int qt = task & 127;
    const int h  = (task >> 7) & 15;
    const int b  = task >> 11;
    const int q0 = qt * 16;

    const bf16* Qp = q  + ((size_t)b * H_ + h) * (size_t)T_ * D_;
    const bf16* Kp = k  + ((size_t)b * H_ + h) * (size_t)T_ * D_;
    const bf16* Vp = vt + ((size_t)b * H_ + h) * (size_t)D_ * T_;

    // Q A-fragments: rows q0..q0+15, d in [0,32) and [32,64)
    v16bf qa[2];
    #pragma unroll
    for (int f = 0; f < 2; ++f) {
        const bf16* p = Qp + (size_t)(q0 + l16) * D_ + f * 32 + 8 * hl;
        v8bf lo = *reinterpret_cast<const v8bf*>(p);
        v8bf hi = *reinterpret_cast<const v8bf*>(p + 16);
        #pragma unroll
        for (int e = 0; e < 8; ++e) { qa[f][e] = lo[e]; qa[f][e + 8] = hi[e]; }
    }

    v8f o[4] = {};
    float mrun[8], lrun[8];
    #pragma unroll
    for (int r = 0; r < 8; ++r) { mrun[r] = -1e30f; lrun[r] = 0.0f; }

    const float scale = 0.125f;   // 1/sqrt(64)

    for (int kc = 0; kc < q0 + 16; kc += 32) {
        // prefetch next chunk's K rows and V columns
        if (kc + 32 < q0 + 16) {
            __builtin_prefetch(Kp + (size_t)(kc + 32 + l16) * D_, 0, 3);
            __builtin_prefetch(Kp + (size_t)(kc + 48 + l16) * D_, 0, 3);
            __builtin_prefetch(Vp + (size_t)l16 * T_ + kc + 32, 0, 3);
        }

        v8f s[2] = {};
        #pragma unroll
        for (int ni = 0; ni < 2; ++ni) {
            #pragma unroll
            for (int f = 0; f < 2; ++f) {
                const bf16* p = Kp + (size_t)(kc + ni * 16 + l16) * D_ + f * 32 + 8 * hl;
                v8bf lo = *reinterpret_cast<const v8bf*>(p);
                v8bf hi = *reinterpret_cast<const v8bf*>(p + 16);
                v16bf kb;
                #pragma unroll
                for (int e = 0; e < 8; ++e) { kb[e] = lo[e]; kb[e + 8] = hi[e]; }
                s[ni] = __builtin_amdgcn_wmma_f32_16x16x32_bf16(
                    false, qa[f], false, kb, (short)0, s[ni], false, false);
            }
        }

        // scale + causal mask + chunk row-max
        float mc[8];
        #pragma unroll
        for (int r = 0; r < 8; ++r) {
            const int row = q0 + r + 8 * hl;
            #pragma unroll
            for (int ni = 0; ni < 2; ++ni) {
                const int col = kc + ni * 16 + l16;
                float v = s[ni][r] * scale;
                if (col > row) v = -1e30f;
                s[ni][r] = v;
            }
            mc[r] = fmaxf(s[0][r], s[1][r]);
        }
        #pragma unroll
        for (int off = 1; off < 16; off <<= 1)
            #pragma unroll
            for (int r = 0; r < 8; ++r)
                mc[r] = fmaxf(mc[r], __shfl_xor(mc[r], off, 32));

        float alpha[8], rs[8];
        #pragma unroll
        for (int r = 0; r < 8; ++r) {
            const float mnew = fmaxf(mrun[r], mc[r]);
            alpha[r] = __expf(mrun[r] - mnew);
            mrun[r]  = mnew;
            const float p0 = __expf(s[0][r] - mnew);
            const float p1 = __expf(s[1][r] - mnew);
            s[0][r] = p0; s[1][r] = p1;
            rs[r] = p0 + p1;
        }
        #pragma unroll
        for (int off = 1; off < 16; off <<= 1)
            #pragma unroll
            for (int r = 0; r < 8; ++r)
                rs[r] += __shfl_xor(rs[r], off, 32);
        #pragma unroll
        for (int r = 0; r < 8; ++r) lrun[r] = lrun[r] * alpha[r] + rs[r];
        #pragma unroll
        for (int j = 0; j < 4; ++j)
            #pragma unroll
            for (int r = 0; r < 8; ++r)
                o[j][r] *= alpha[r];

        // P: C-layout -> A-layout through per-wave LDS (same-wave DS is in-order)
        bf16* st = &pstage[wave][0];
        #pragma unroll
        for (int ni = 0; ni < 2; ++ni)
            #pragma unroll
            for (int r = 0; r < 8; ++r)
                st[(r + 8 * hl) * LDA_P + ni * 16 + l16] = f2bf(s[ni][r]);

        v16bf pa;
        {
            const bf16* p = st + l16 * LDA_P + 8 * hl;
            v8bf lo = *reinterpret_cast<const v8bf*>(p);
            v8bf hi = *reinterpret_cast<const v8bf*>(p + 16);
            #pragma unroll
            for (int e = 0; e < 8; ++e) { pa[e] = lo[e]; pa[e + 8] = hi[e]; }
        }

        // O += P @ V  (V^T rows are contiguous in t)
        #pragma unroll
        for (int j = 0; j < 4; ++j) {
            const bf16* p = Vp + (size_t)(j * 16 + l16) * T_ + kc + 8 * hl;
            v8bf lo = *reinterpret_cast<const v8bf*>(p);
            v8bf hi = *reinterpret_cast<const v8bf*>(p + 16);
            v16bf vb;
            #pragma unroll
            for (int e = 0; e < 8; ++e) { vb[e] = lo[e]; vb[e + 8] = hi[e]; }
            o[j] = __builtin_amdgcn_wmma_f32_16x16x32_bf16(
                false, pa, false, vb, (short)0, o[j], false, false);
        }
    }

    // normalize + write y[b][t][h*64+d] (bf16, merged heads)
    #pragma unroll
    for (int r = 0; r < 8; ++r) {
        const float inv = 1.0f / lrun[r];
        const int t = q0 + r + 8 * hl;
        #pragma unroll
        for (int j = 0; j < 4; ++j) {
            const int c = h * D_ + j * 16 + l16;
            y[((size_t)b * T_ + t) * C_ + c] = f2bf(o[j][r] * inv);
        }
    }
}

// ---------------------------------------------------------------------------
// GEMM: out = y_bf @ w_proj_bf  (f32 output)
// ---------------------------------------------------------------------------
__global__ __launch_bounds__(256) void gemm_proj_kernel(
    const bf16* __restrict__ A,    // [8192, 1024]
    const bf16* __restrict__ Bw,   // [1024, 1024]
    float* __restrict__ out)       // [8192, 1024]
{
    __shared__ __align__(16) bf16 As[128 * LDA_P];
    __shared__ __align__(16) bf16 Bt[128 * LDB_P];

    const int lane = threadIdx.x & 31;
    const int wave = threadIdx.x >> 5;
    const int hl   = lane >> 4;
    const int l16  = lane & 15;

    const int nTiles = C_ / 128;                  // 8
    const int mBase  = (blockIdx.x / nTiles) * 128;
    const int nBase  = (blockIdx.x % nTiles) * 128;

    v8f acc[2][4] = {};
    gemm_mainloop<C_>(A, Bw, As, Bt, mBase, nBase, acc);

    const int mSub = (wave & 3) * 32;
    const int nSub = (wave >> 2) * 64;

    #pragma unroll
    for (int mi = 0; mi < 2; ++mi) {
        #pragma unroll
        for (int ni = 0; ni < 4; ++ni) {
            const int col = nBase + nSub + ni * 16 + l16;
            #pragma unroll
            for (int r = 0; r < 8; ++r) {
                const int row = mBase + mSub + mi * 16 + r + 8 * hl;
                out[(size_t)row * C_ + col] = acc[mi][ni][r];
            }
        }
    }
}

// ---------------------------------------------------------------------------
// Launch
// ---------------------------------------------------------------------------
extern "C" void kernel_launch(void* const* d_in, const int* in_sizes, int n_in,
                              void* d_out, int out_size, void* d_ws, size_t ws_size,
                              hipStream_t stream) {
    (void)in_sizes; (void)n_in; (void)out_size; (void)ws_size;

    const float* x      = (const float*)d_in[0];   // [4,2048,1024]
    const float* w_qkv  = (const float*)d_in[1];   // [1024,3072]
    const float* w_proj = (const float*)d_in[2];   // [1024,1024]
    float* out = (float*)d_out;                    // [4,2048,1024]

    char* ws = (char*)d_ws;
    size_t off = 0;
    bf16* x_bf     = (bf16*)(ws + off); off += (size_t)8192 * 1024 * 2;  // 16 MB
    bf16* wqkv_bf  = (bf16*)(ws + off); off += (size_t)1024 * 3072 * 2;  //  6 MB
    bf16* wproj_bf = (bf16*)(ws + off); off += (size_t)1024 * 1024 * 2;  //  2 MB
    bf16* qb       = (bf16*)(ws + off); off += (size_t)8192 * 1024 * 2;  // 16 MB  [B,H,T,D]
    bf16* kb       = (bf16*)(ws + off); off += (size_t)8192 * 1024 * 2;  // 16 MB  [B,H,T,D]
    bf16* vtb      = (bf16*)(ws + off); off += (size_t)8192 * 1024 * 2;  // 16 MB  [B,H,D,T]
    bf16* yb       = (bf16*)(ws + off); off += (size_t)8192 * 1024 * 2;  // 16 MB  [B,T,C]

    {
        int n4 = 8192 * 1024 / 4;
        cvt4_kernel<<<(n4 + 255) / 256, 256, 0, stream>>>(x, x_bf, n4);
    }
    {
        int n4 = 1024 * 3072 / 4;
        cvt4_kernel<<<(n4 + 255) / 256, 256, 0, stream>>>(w_qkv, wqkv_bf, n4);
    }
    {
        int n4 = 1024 * 1024 / 4;
        cvt4_kernel<<<(n4 + 255) / 256, 256, 0, stream>>>(w_proj, wproj_bf, n4);
    }

    gemm_qkv_kernel<<<64 * 24, 256, 0, stream>>>(x_bf, wqkv_bf, qb, kb, vtb);
    attn_kernel<<<1024, 256, 0, stream>>>(qb, kb, vtb, yb);
    gemm_proj_kernel<<<64 * 8, 256, 0, stream>>>(yb, wproj_bf, out);
}
